// ROI_Feature_45174466019465
// MI455X (gfx1250) — matrix operand order repaired and checked by reading, not compiled, not run on hardware
//
#include <hip/hip_runtime.h>

typedef __attribute__((ext_vector_type(16))) _Float16 v16h;
typedef __attribute__((ext_vector_type(8)))  float    v8f;
typedef __attribute__((ext_vector_type(2)))  float    v2f;

// ---- fixed problem sizes (from setup_inputs) ----
#define NB 16
#define NK 16
#define NN 256
#define NP 240
#define CF 1024
#define FH 38
#define FW 38
#define KDIST 36
#define CEMB 200
#define CPOS 128
#define CFEAT 2048
#define COBJ 2376
#define MSZ 27
#define NTILE 13            // ceil(200/16)

// ---- d_out offsets (float elements, reference return order) ----
#define OFF_OBJF    0
#define OFF_UFEAT   608256
#define OFF_UBOX    12650496
#define OFF_SPATIAL 12651696
#define OFF_PAIR    13001616
#define OFF_IMIDX   13002096
#define OFF_LABELS  13002336
#define OFF_SCORES  13002592

// ============================================================
// Kernel 1: per-batch human argmax, pair construction, label/score
// edits, scaled union boxes. One block, 256 threads.
// ============================================================
__global__ void meta_kernel(const float* __restrict__ boxes4,
                            const float* __restrict__ dist,
                            const int*   __restrict__ labels_in,
                            const float* __restrict__ scores_in,
                            const float* __restrict__ im_info,
                            float* __restrict__ ubox_out,
                            float* __restrict__ pair_out,
                            float* __restrict__ imidx_out,
                            float* __restrict__ labels_out,
                            float* __restrict__ scores_out,
                            int*   __restrict__ pair_ws) {
  __shared__ int s_h[NB];
  int t = threadIdx.x;
  if (t < NB) {
    int base = t * NK;
    float best = dist[(size_t)base * KDIST];
    int bi = 0;
    for (int k = 1; k < NK; ++k) {
      float v = dist[(size_t)(base + k) * KDIST];
      if (v > best) { best = v; bi = k; }   // first-max like jnp.argmax
    }
    s_h[t] = base + bi;
  }
  __syncthreads();

  // pred_labels / pred_scores (256 entries)
  {
    int i = t;
    int h = s_h[i >> 4];
    bool isH = (i == h);
    labels_out[i] = isH ? 0.0f : (float)labels_in[i];
    scores_out[i] = isH ? dist[(size_t)i * KDIST] : scores_in[i];
  }

  // pairs + im_idx + union boxes (240 entries)
  if (t < NP) {
    int b = t / (NK - 1), j = t % (NK - 1);
    int h = s_h[b];
    int hl = h - b * NK;
    int ol = (j < hl) ? j : j + 1;   // j-th non-human local index (ascending)
    int o = b * NK + ol;
    pair_out[t * 2 + 0] = (float)h;
    pair_out[t * 2 + 1] = (float)o;
    pair_ws[t * 2 + 0] = h;
    pair_ws[t * 2 + 1] = o;
    imidx_out[t] = (float)b;
    float sc = im_info[0];
    float hx1 = boxes4[h*4+0]*sc, hy1 = boxes4[h*4+1]*sc;
    float hx2 = boxes4[h*4+2]*sc, hy2 = boxes4[h*4+3]*sc;
    float ox1 = boxes4[o*4+0]*sc, oy1 = boxes4[o*4+1]*sc;
    float ox2 = boxes4[o*4+2]*sc, oy2 = boxes4[o*4+3]*sc;
    ubox_out[t*5+0] = (float)b;
    ubox_out[t*5+1] = fminf(hx1, ox1);
    ubox_out[t*5+2] = fminf(hy1, oy1);
    ubox_out[t*5+3] = fmaxf(hx2, ox2);
    ubox_out[t*5+4] = fmaxf(hy2, oy2);
  }
}

// ============================================================
// Kernel 2a: copy features (256x2048) into obj_features[:, 0:2048]
// row stride 2376 floats = 9504 B (16B aligned) -> float4 path.
// ============================================================
__global__ void features_copy_kernel(const float* __restrict__ features,
                                     float* __restrict__ objf) {
  int row = blockIdx.x;
  const float4* src = (const float4*)(features + (size_t)row * CFEAT);
  float4* dst = (float4*)(objf + (size_t)row * COBJ);
  for (int i = threadIdx.x; i < CFEAT / 4; i += blockDim.x)
    dst[i] = src[i];
}

// ============================================================
// Kernel 2b: obj_embed = dist(256x36) @ embed_w(36x200) via
// v_wmma_f32_16x16x32_f16 (f32 accumulate).
// Block = 256 threads handles one 16-row tile; operands are staged
// in LDS *in fragment order* (zero-padded to K=64, N=208) so each
// wave's operand read is two contiguous 32-byte LDS loads -> no
// divergent scalar global loads in front of the WMMA.
// Writes obj_features[:, 2048:2248].
// ============================================================
__global__ void objembed_wmma_kernel(const float* __restrict__ dist,
                                     const float* __restrict__ embed_w,
                                     float* __restrict__ objf) {
  // fragment-ordered LDS staging
  __shared__ alignas(32) _Float16 sA[2][32][16];          // [ks][lane][j]  (2 KB)
  __shared__ alignas(32) _Float16 sB[NTILE][2][32][16];   // [tn][ks][lane][j] (26 KB)

  int tm = blockIdx.x;            // 16 row tiles
  int t = threadIdx.x;

  // ---- fill A fragments: 2*32*16 = 1024 halves, 4 iters ----
  for (int idx = t; idx < 2 * 32 * 16; idx += 256) {
    int j    = idx & 15;
    int lane = (idx >> 4) & 31;
    int ks   = idx >> 9;
    int hi   = lane >> 4;
    int row  = lane & 15;
    // A 16x32 f16 layout: VGPRs 0-3 pack K 2i,2i+1 (lanes16-31: +8),
    // VGPRs 4-7 pack K 16.. (lanes16-31: +8)
    int ka = ks * 32 + ((j < 8) ? (j + hi * 8) : (j + 8 + hi * 8));
    int m  = tm * 16 + row;
    float v = dist[(size_t)m * KDIST + min(ka, KDIST - 1)];  // clamped, in-bounds
    if (ka >= KDIST) v = 0.0f;                               // select, no branch
    sA[ks][lane][j] = (_Float16)v;
  }

  // ---- fill B fragments: 13*2*32*16 = 13312 halves, 52 iters ----
  for (int idx = t; idx < NTILE * 2 * 32 * 16; idx += 256) {
    int j    = idx & 15;
    int lane = (idx >> 4) & 31;
    int ks   = (idx >> 9) & 1;
    int tn   = idx >> 10;
    int hi   = lane >> 4;
    // B 32x16 f16 layout: same K order as A (j<8: K j+8*hi; j>=8: K 16+8*hi+(j-8))
    int kb = ks * 32 + ((j < 8) ? (j + hi * 8) : (j + 8 + hi * 8));
    int n  = tn * 16 + (lane & 15);
    float v = embed_w[(size_t)min(kb, KDIST - 1) * CEMB + min(n, CEMB - 1)];
    if (kb >= KDIST || n >= CEMB) v = 0.0f;
    sB[tn][ks][lane][j] = (_Float16)v;
  }
  __syncthreads();

  int wave = t >> 5;
  int lane = t & 31;
  int laneLo = lane & 15;
  int laneHi = lane >> 4;

  for (int tn = wave; tn < NTILE; tn += 8) {
    v8f acc = {};
    #pragma unroll
    for (int ks = 0; ks < 2; ++ks) {
      v16h a = *(const v16h*)(&sA[ks][lane][0]);       // 32B contiguous LDS read
      v16h b = *(const v16h*)(&sB[tn][ks][lane][0]);   // 32B contiguous LDS read
      acc = __builtin_amdgcn_wmma_f32_16x16x32_f16(
          false, a, false, b, (short)0, acc, false, false);
    }
    int n = tn * 16 + laneLo;
    #pragma unroll
    for (int v = 0; v < 8; ++v) {
      int mm = tm * 16 + v + laneHi * 8;
      if (n < CEMB)
        objf[(size_t)mm * COBJ + CFEAT + n] = acc[v];
    }
  }
}

// ============================================================
// Kernel 2c: pos_embed = relu(xb(256x4) @ lin_w(4x128) + lin_b)
// xb = batchnorm(center_size(boxes4)). K=4 -> exact fit for
// V_WMMA_F32_16X16X4_F32. One wave per 16x16 tile (16x8 tiles).
// ============================================================
__device__ __forceinline__ float xb_val(const float* __restrict__ boxes4,
                                        const float* __restrict__ g,
                                        const float* __restrict__ bt,
                                        const float* __restrict__ mu,
                                        const float* __restrict__ var,
                                        int m, int k) {
  const float* bx = boxes4 + (size_t)m * 4;
  float cs;
  if (k == 0)      cs = (bx[0] + bx[2]) * 0.5f;
  else if (k == 1) cs = (bx[1] + bx[3]) * 0.5f;
  else if (k == 2) cs = bx[2] - bx[0];
  else             cs = bx[3] - bx[1];
  return (cs - mu[k]) * rsqrtf(var[k] + 1e-5f) * g[k] + bt[k];
}

__global__ void posembed_wmma_kernel(const float* __restrict__ boxes4,
                                     const float* __restrict__ bn_gamma,
                                     const float* __restrict__ bn_beta,
                                     const float* __restrict__ bn_mean,
                                     const float* __restrict__ bn_var,
                                     const float* __restrict__ lin_w,
                                     const float* __restrict__ lin_b,
                                     float* __restrict__ objf) {
  int tile = blockIdx.x;          // 0..127
  int tm = tile & 15;             // 16 row tiles
  int tn = tile >> 4;             // 8 col tiles
  int lane = threadIdx.x & 31;
  int laneLo = lane & 15;
  int laneHi = lane >> 4;
  int n = tn * 16 + laneLo;

  v8f acc = {};
#if __has_builtin(__builtin_amdgcn_wmma_f32_16x16x4_f32)
  {
    // A 16x4 f32: lanes 0-15 -> K0,K1 ; lanes 16-31 -> K2,K3
    // B 4x16 f32: lanes 0-15 -> rows K0,K1 ; lanes 16-31 -> rows K2,K3
    int m = tm * 16 + laneLo;
    int k0 = laneHi * 2;
    v2f a, b;
    #pragma unroll
    for (int tk = 0; tk < 2; ++tk) {
      int k = k0 + tk;
      a[tk] = xb_val(boxes4, bn_gamma, bn_beta, bn_mean, bn_var, m, k);
      b[tk] = lin_w[(size_t)k * CPOS + n];
    }
    acc = __builtin_amdgcn_wmma_f32_16x16x4_f32(
        false, a, false, b, (short)0, acc, false, false);
  }
#else
  // Scalar fallback: each lane computes its 8 outputs directly (K=4 dot).
  #pragma unroll
  for (int v = 0; v < 8; ++v) {
    int mm = tm * 16 + v + laneHi * 8;
    float s = 0.0f;
    #pragma unroll
    for (int k = 0; k < 4; ++k)
      s += xb_val(boxes4, bn_gamma, bn_beta, bn_mean, bn_var, mm, k) *
           lin_w[(size_t)k * CPOS + n];
    acc[v] = s;
  }
#endif
  float bias = lin_b[n];
  #pragma unroll
  for (int v = 0; v < 8; ++v) {
    int mm = tm * 16 + v + laneHi * 8;
    objf[(size_t)mm * COBJ + CFEAT + CEMB + n] = fmaxf(acc[v] + bias, 0.0f);
  }
}

// ============================================================
// Kernel 3: roi_align over union boxes.
// grid = (240 pairs, 16 channel-tiles of 64), block = 256.
// Bilinear sample coordinates/weights computed once in LDS.
// fmaps (94.6 MB) fits in the 192 MB L2 -> direct gathers are cheap.
// ============================================================
__global__ void roi_align_kernel(const float* __restrict__ fmaps,
                                 const float* __restrict__ ubox,
                                 float* __restrict__ ufeat) {
  __shared__ float s_lx[14], s_hx[14], s_ly[14], s_hy[14];
  __shared__ int   s_x0[14], s_x1[14], s_y0[14], s_y1[14];
  int p = blockIdx.x;          // pair
  int ct = blockIdx.y;         // channel tile (64 channels)
  int t = threadIdx.x;

  if (t < 28) {
    bool isY = (t >= 14);
    int i = isY ? t - 14 : t;
    const float* ub = ubox + (size_t)p * 5;
    float c1 = ub[isY ? 2 : 1] * (1.0f / 16.0f);
    float c2 = ub[isY ? 4 : 3] * (1.0f / 16.0f);
    float r = fmaxf(c2 - c1, 1.0f);
    float s = c1 + r * ((i + 0.5f) / 14.0f);
    s = fminf(fmaxf(s, 0.0f), (float)(FW - 1));   // H == W == 38
    int i0 = (int)floorf(s);
    int i1 = min(i0 + 1, FW - 1);
    float l = s - (float)i0;
    if (isY) { s_y0[i] = i0; s_y1[i] = i1; s_ly[i] = l; s_hy[i] = 1.0f - l; }
    else     { s_x0[i] = i0; s_x1[i] = i1; s_lx[i] = l; s_hx[i] = 1.0f - l; }
  }
  __syncthreads();

  int b = p / (NK - 1);
  const float* fbase = fmaps + ((size_t)b * CF + (size_t)ct * 64) * (FH * FW);
  float* obase = ufeat + ((size_t)p * CF + (size_t)ct * 64) * 49;

  // Warm the channel tile (global_prefetch_b8); 64ch*1444*4B spread over lanes.
  __builtin_prefetch(fbase + (size_t)t * (64 * FH * FW / 256), 0, 0);

  for (int item = t; item < 64 * 49; item += 256) {
    int c = item / 49;
    int pix = item - c * 49;
    int oy = pix / 7, ox = pix - oy * 7;
    const float* f = fbase + (size_t)c * (FH * FW);
    float acc = 0.0f;
    #pragma unroll
    for (int sy = 0; sy < 2; ++sy) {
      int iy = oy * 2 + sy;
      const float* r0 = f + s_y0[iy] * FW;
      const float* r1 = f + s_y1[iy] * FW;
      float hy = s_hy[iy], ly = s_ly[iy];
      #pragma unroll
      for (int sx = 0; sx < 2; ++sx) {
        int ix = ox * 2 + sx;
        int x0 = s_x0[ix], x1 = s_x1[ix];
        float hx = s_hx[ix], lx = s_lx[ix];
        acc += (r0[x0] * hx + r0[x1] * lx) * hy +
               (r1[x0] * hx + r1[x1] * lx) * ly;
      }
    }
    obase[item] = acc * 0.25f;
  }
}

// ============================================================
// Kernel 4: spatial masks (240 x 2 x 27 x 27), value in {+0.5,-0.5}.
// ============================================================
__global__ void spatial_mask_kernel(const float* __restrict__ boxes4,
                                    const int* __restrict__ pair_ws,
                                    float* __restrict__ out) {
  int p = blockIdx.x;
  int h = pair_ws[p * 2 + 0];
  int o = pair_ws[p * 2 + 1];
  float sx1 = boxes4[h*4+0], sy1 = boxes4[h*4+1], sx2 = boxes4[h*4+2], sy2 = boxes4[h*4+3];
  float ox1 = boxes4[o*4+0], oy1 = boxes4[o*4+1], ox2 = boxes4[o*4+2], oy2 = boxes4[o*4+3];
  float ux1 = fminf(sx1, ox1), uy1 = fminf(sy1, oy1);
  float ux2 = fmaxf(sx2, ox2), uy2 = fmaxf(sy2, oy2);
  float iw = (float)MSZ / fmaxf(ux2 - ux1, 1e-6f);
  float ih = (float)MSZ / fmaxf(uy2 - uy1, 1e-6f);

  for (int item = threadIdx.x; item < 2 * MSZ * MSZ; item += blockDim.x) {
    int which = item / (MSZ * MSZ);
    int rem = item - which * (MSZ * MSZ);
    int gy = rem / MSZ, gx = rem - gy * MSZ;
    float bx1 = which ? ox1 : sx1, bx2 = which ? ox2 : sx2;
    float by1 = which ? oy1 : sy1, by2 = which ? oy2 : sy2;
    float fx1 = (bx1 - ux1) * iw, fx2 = (bx2 - ux1) * iw;
    float fy1 = (by1 - uy1) * ih, fy2 = (by2 - uy1) * ih;
    float gxx = gx + 0.5f, gyy = gy + 0.5f;
    bool inside = (gxx >= fx1) && (gxx <= fx2) && (gyy >= fy1) && (gyy <= fy2);
    out[(size_t)p * (2 * MSZ * MSZ) + item] = (inside ? 1.0f : 0.0f) - 0.5f;
  }
}

// ============================================================
extern "C" void kernel_launch(void* const* d_in, const int* in_sizes, int n_in,
                              void* d_out, int out_size, void* d_ws, size_t ws_size,
                              hipStream_t stream) {
  // inputs (setup_inputs order); d_in[0] = box_idx (unused by reference body)
  const float* boxes4    = (const float*)d_in[1];
  const float* features  = (const float*)d_in[2];
  const float* dist      = (const float*)d_in[3];
  const int*   labels_in = (const int*)  d_in[4];
  const float* scores_in = (const float*)d_in[5];
  const float* im_info   = (const float*)d_in[6];
  const float* fmaps     = (const float*)d_in[7];
  const float* embed_w   = (const float*)d_in[8];
  const float* bn_gamma  = (const float*)d_in[9];
  const float* bn_beta   = (const float*)d_in[10];
  const float* bn_mean   = (const float*)d_in[11];
  const float* bn_var    = (const float*)d_in[12];
  const float* lin_w     = (const float*)d_in[13];
  const float* lin_b     = (const float*)d_in[14];

  float* out = (float*)d_out;
  int*   pair_ws = (int*)d_ws;   // 240*2 ints

  meta_kernel<<<1, 256, 0, stream>>>(
      boxes4, dist, labels_in, scores_in, im_info,
      out + OFF_UBOX, out + OFF_PAIR, out + OFF_IMIDX,
      out + OFF_LABELS, out + OFF_SCORES, pair_ws);

  features_copy_kernel<<<NN, 256, 0, stream>>>(features, out + OFF_OBJF);

  objembed_wmma_kernel<<<NN / 16, 256, 0, stream>>>(dist, embed_w, out + OFF_OBJF);

  posembed_wmma_kernel<<<16 * 8, 32, 0, stream>>>(
      boxes4, bn_gamma, bn_beta, bn_mean, bn_var, lin_w, lin_b, out + OFF_OBJF);

  roi_align_kernel<<<dim3(NP, CF / 64), 256, 0, stream>>>(
      fmaps, out + OFF_UBOX, out + OFF_UFEAT);

  spatial_mask_kernel<<<NP, 256, 0, stream>>>(boxes4, pair_ws, out + OFF_SPATIAL);
}
